// MultiHeadAttention_23914377904179
// MI455X (gfx1250) — matrix operand order repaired and checked by reading, not compiled, not run on hardware
//
#include <hip/hip_runtime.h>
#include <hip/hip_bf16.h>

#define HID   2048
#define NH    16
#define HD    128
#define BATCH 2
#define SEQ   2048
#define MROWS (BATCH * SEQ)   // 4096

typedef __attribute__((ext_vector_type(16))) __bf16 v16bf;
typedef __attribute__((ext_vector_type(8)))  __bf16 v8bf;
typedef __attribute__((ext_vector_type(8)))  float  v8f;

__device__ __forceinline__ v8f wmma_bf16(v16bf a, v16bf b, v8f c) {
  // D = A(16x32 bf16) * B(32x16 bf16) + C(16x16 f32)
  return __builtin_amdgcn_wmma_f32_16x16x32_bf16(false, a, false, b, (short)0, c,
                                                 false, false);
}

// XOR-lane shuffle via ds_swizzle (group-of-32 mode: offset = xor<<10 | and 0x1f).
// Immediate-encoded: no index VGPRs, no VALU address math.
template <int XMASK>
__device__ __forceinline__ float swz_xor(float v) {
  int r = __builtin_amdgcn_ds_swizzle(__float_as_int(v), (XMASK << 10) | 0x1f);
  return __int_as_float(r);
}

// max-reduce across the 16 lanes of a half-wave (masks 1,2,4,8 stay in-half)
__device__ __forceinline__ float halfmax16(float v) {
  v = fmaxf(v, swz_xor<1>(v));
  v = fmaxf(v, swz_xor<2>(v));
  v = fmaxf(v, swz_xor<4>(v));
  v = fmaxf(v, swz_xor<8>(v));
  return v;
}

// ---------------------------------------------------------------------------
// fp32 -> bf16 conversion, 8 elements / thread (n must be a multiple of 8)
// ---------------------------------------------------------------------------
__global__ void f32_to_bf16_k(const float* __restrict__ in,
                              __bf16* __restrict__ out, int n) {
  int i = (blockIdx.x * blockDim.x + threadIdx.x) * 8;
  if (i + 7 < n) {
    float4 a = *(const float4*)(in + i);
    float4 b = *(const float4*)(in + i + 4);
    v8bf o;
    o[0] = (__bf16)a.x; o[1] = (__bf16)a.y; o[2] = (__bf16)a.z; o[3] = (__bf16)a.w;
    o[4] = (__bf16)b.x; o[5] = (__bf16)b.y; o[6] = (__bf16)b.z; o[7] = (__bf16)b.w;
    *(v8bf*)(out + i) = o;
  }
}

// ---------------------------------------------------------------------------
// V [B*S, HID] (bf16) -> Vt [B, NH, HD, SEQ] (bf16)
// ---------------------------------------------------------------------------
__global__ void transpose_v_k(const __bf16* __restrict__ v,
                              __bf16* __restrict__ vt) {
  long i = (long)blockIdx.x * blockDim.x + threadIdx.x;
  if (i >= (long)MROWS * HID) return;
  int  col = (int)(i % HID);
  long row = i / HID;                  // row = b*SEQ + s
  int  b = (int)(row / SEQ), s = (int)(row % SEQ);
  int  h = col / HD, d = col % HD;
  vt[((long)(b * NH + h) * HD + d) * SEQ + s] = v[i];
}

// ---------------------------------------------------------------------------
// NT GEMM:  C[M,N] = (A[M,K] * Bw[N,K]^T + bias[N]) * scale
// A, Bw bf16 row-major (K contiguous).  One wave = 64x64 tile (4x4 WMMA tiles).
// ---------------------------------------------------------------------------
template <int OUT_F32>
__global__ __launch_bounds__(256, 1)
void gemm_nt_bf16(const __bf16* __restrict__ A,
                  const __bf16* __restrict__ Bw,
                  const float* __restrict__ bias,
                  void* __restrict__ Cout,
                  int Mm, int Nn, int Kk, float scale) {
  const int lane = threadIdx.x & 31;
  const int wave = threadIdx.x >> 5;
  const int l16  = lane & 15;
  const int hf   = lane >> 4;          // half-wave select
  const int tilesN = Nn >> 6;
  const int wg = blockIdx.x * (blockDim.x >> 5) + wave;
  const int m0 = (wg / tilesN) * 64;
  const int n0 = (wg % tilesN) * 64;
  if (m0 >= Mm) return;

  v8f acc[4][4] = {};

  for (int k0 = 0; k0 < Kk; k0 += 32) {
    v16bf af[4], bfr[4];
#pragma unroll
    for (int i = 0; i < 4; ++i) {
      // A 16x32 bf16 fragment, contiguous k order used here: lanes 0-15 row M=l16 K[0..15];
      //                      lanes 16-31 same row  K[16..31]
      const __bf16* p = A + (size_t)(m0 + 16 * i + l16) * Kk + k0 + 16 * hf;
      v8bf lo = *(const v8bf*)p;
      v8bf hi = *(const v8bf*)(p + 8);
#pragma unroll
      for (int t = 0; t < 8; ++t) { af[i][t] = lo[t]; af[i][t + 8] = hi[t]; }
    }
#pragma unroll
    for (int j = 0; j < 4; ++j) {
      // B 32x16 bf16 fragment, same k order as A: lane column N=l16; halves = 16 consecutive K
      const __bf16* p = Bw + (size_t)(n0 + 16 * j + l16) * Kk + k0 + 16 * hf;
      bfr[j] = *(const v16bf*)p;
    }
#pragma unroll
    for (int i = 0; i < 4; ++i)
#pragma unroll
      for (int j = 0; j < 4; ++j)
        acc[i][j] = wmma_bf16(af[i], bfr[j], acc[i][j]);
  }

  // Epilogue.  C/D layout: lane column N=n0+16j+l16, VGPR r -> row r + 8*hf.
#pragma unroll
  for (int j = 0; j < 4; ++j) {
    const int n = n0 + 16 * j + l16;
    const float bb = bias ? bias[n] : 0.0f;
#pragma unroll
    for (int i = 0; i < 4; ++i) {
#pragma unroll
      for (int r = 0; r < 8; ++r) {
        const int m = m0 + 16 * i + r + 8 * hf;
        const float v = (acc[i][j][r] + bb) * scale;
        if (OUT_F32) ((float*)Cout)[(size_t)m * Nn + n] = v;
        else         ((__bf16*)Cout)[(size_t)m * Nn + n] = (__bf16)v;
      }
    }
  }
}

// ---------------------------------------------------------------------------
// Causal flash attention.  One wave = 16 query rows of one (batch, head).
// Q pre-scaled by (1/sqrt(d))*log2(e) so softmax uses exp2 (v_exp_f32).
// Row sums are accumulated with an extra WMMA against an all-ones B fragment
// (P row-sum == P @ ones), so only the running-max needs lane shuffles.
// ---------------------------------------------------------------------------
__global__ __launch_bounds__(256, 1)
void flash_attn_k(const __bf16* __restrict__ Q,
                  const __bf16* __restrict__ Kc,
                  const __bf16* __restrict__ Vt,
                  __bf16* __restrict__ Ctx) {
  __shared__ __align__(16) __bf16 pls[8][16][40];  // per-wave 16x32 P tile, padded

  const int lane = threadIdx.x & 31;
  const int wave = threadIdx.x >> 5;
  const int l16  = lane & 15;
  const int hf   = lane >> 4;
  const int wg = blockIdx.x * (blockDim.x >> 5) + wave;
  const int qt = wg % (SEQ / 16);
  const int bh = wg / (SEQ / 16);
  const int h = bh % NH, b = bh / NH;
  const int q0 = qt * 16;

  // Q fragments over the 128-wide head dim (4 x K=32), resident in VGPRs
  v16bf qa[4];
  {
    const __bf16* qrow = Q + (size_t)(b * SEQ + q0 + l16) * HID + h * HD;
#pragma unroll
    for (int kc = 0; kc < 4; ++kc) {
      const __bf16* p = qrow + kc * 32 + 16 * hf;
      v8bf lo = *(const v8bf*)p;
      v8bf hi = *(const v8bf*)(p + 8);
#pragma unroll
      for (int t = 0; t < 8; ++t) { qa[kc][t] = lo[t]; qa[kc][t + 8] = hi[t]; }
    }
  }

  // all-ones bf16 B fragment (row-sum trick)
  v16bf ones;
#pragma unroll
  for (int t = 0; t < 16; ++t) ones[t] = (__bf16)1.0f;

  v8f o[8] = {};
  v8f ls = {};            // running row-sum accumulator (every column identical)
  float mrun[8];
#pragma unroll
  for (int r = 0; r < 8; ++r) mrun[r] = -3.0e38f;

  const __bf16* kbase = Kc + (size_t)b * SEQ * HID + h * HD;
  const __bf16* vbase = Vt + (size_t)(b * NH + h) * HD * SEQ;
  const int npair = (q0 + 16 + 31) >> 5;  // key pair-tiles needed (causal)

  for (int kp = 0; kp < npair; ++kp) {
    const int kb0 = kp * 32;
    v8f s0 = {}, s1 = {};
#pragma unroll
    for (int kc = 0; kc < 4; ++kc) {
      v16bf kf0 = *(const v16bf*)(kbase + (size_t)(kb0 + l16) * HID + kc * 32 + 16 * hf);
      v16bf kf1 = *(const v16bf*)(kbase + (size_t)(kb0 + 16 + l16) * HID + kc * 32 + 16 * hf);
      s0 = wmma_bf16(qa[kc], kf0, s0);
      s1 = wmma_bf16(qa[kc], kf1, s1);
    }

    if (kb0 + 31 > q0) {  // tile touches the diagonal -> causal mask
#pragma unroll
      for (int r = 0; r < 8; ++r) {
        const int q = q0 + r + 8 * hf;
        if (kb0 + l16 > q)      s0[r] = -3.0e38f;
        if (kb0 + 16 + l16 > q) s1[r] = -3.0e38f;
      }
    }

    // online softmax: running max via in-half lane reduction, exp2, rescale
    float alpha[8];
#pragma unroll
    for (int r = 0; r < 8; ++r) {
      const float mx   = halfmax16(fmaxf(s0[r], s1[r]));
      const float mnew = fmaxf(mrun[r], mx);
      alpha[r] = __builtin_amdgcn_exp2f(mrun[r] - mnew);
      s0[r] = __builtin_amdgcn_exp2f(s0[r] - mnew);
      s1[r] = __builtin_amdgcn_exp2f(s1[r] - mnew);
      mrun[r] = mnew;
    }
#pragma unroll
    for (int r = 0; r < 8; ++r) {
      ls[r] *= alpha[r];
#pragma unroll
      for (int j = 0; j < 8; ++j) o[j][r] *= alpha[r];
    }

    // P: C-layout f32 -> LDS bf16 (row-major 16x32) -> A-layout fragment.
    // Same-wave LDS RAW; DScnt ordering keeps this wave-coherent (no barrier).
#pragma unroll
    for (int r = 0; r < 8; ++r) {
      pls[wave][r + 8 * hf][l16]      = (__bf16)s0[r];
      pls[wave][r + 8 * hf][16 + l16] = (__bf16)s1[r];
    }
    v16bf pa;
    {
      const __bf16* p = &pls[wave][l16][16 * hf];
      v8bf lo = *(const v8bf*)p;
      v8bf hi = *(const v8bf*)(p + 8);
#pragma unroll
      for (int t = 0; t < 8; ++t) { pa[t] = lo[t]; pa[t + 8] = hi[t]; }
    }

    // row sums: ls += P @ ones  (one matrix-pipe op instead of 32 shuffles)
    ls = wmma_bf16(pa, ones, ls);

    // O += P(16x32) * V(32x128): Vt rows are d, contiguous along keys (NT)
#pragma unroll
    for (int j = 0; j < 8; ++j) {
      v16bf vf = *(const v16bf*)(vbase + (size_t)(16 * j + l16) * SEQ + kb0 + 16 * hf);
      o[j] = wmma_bf16(pa, vf, o[j]);
    }
  }

  // normalize and store context (bf16, [B*S, HID] layout)
  float rl[8];
#pragma unroll
  for (int r = 0; r < 8; ++r) rl[r] = 1.0f / ls[r];
  __bf16* crow = Ctx + (size_t)(b * SEQ + q0) * HID + h * HD;
#pragma unroll
  for (int j = 0; j < 8; ++j)
#pragma unroll
    for (int r = 0; r < 8; ++r)
      crow[(size_t)(r + 8 * hf) * HID + 16 * j + l16] = (__bf16)(o[j][r] * rl[r]);
}

// ---------------------------------------------------------------------------
extern "C" void kernel_launch(void* const* d_in, const int* in_sizes, int n_in,
                              void* d_out, int out_size, void* d_ws, size_t ws_size,
                              hipStream_t stream) {
  const float* X  = (const float*)d_in[0];
  const float* Wq = (const float*)d_in[1];
  const float* bq = (const float*)d_in[2];
  const float* Wk = (const float*)d_in[3];
  const float* bk = (const float*)d_in[4];
  const float* Wv = (const float*)d_in[5];
  const float* bv = (const float*)d_in[6];
  const float* Wo = (const float*)d_in[7];
  const float* bo = (const float*)d_in[8];
  float* out = (float*)d_out;

  // workspace layout (bytes); total = 128 MiB
  char* ws = (char*)d_ws;
  const size_t SZ_X = (size_t)MROWS * HID * 2;   // 16 MiB
  const size_t SZ_W = (size_t)HID * HID * 2;     //  8 MiB
  __bf16* Xb  = (__bf16*)(ws);
  __bf16* Wqb = (__bf16*)(ws + SZ_X);
  __bf16* Wkb = (__bf16*)(ws + SZ_X + SZ_W);
  __bf16* Wvb = (__bf16*)(ws + SZ_X + 2 * SZ_W);
  __bf16* Wob = (__bf16*)(ws + SZ_X + 3 * SZ_W);
  __bf16* Qb  = (__bf16*)(ws + SZ_X + 4 * SZ_W);
  __bf16* Kb  = (__bf16*)(ws + 2 * SZ_X + 4 * SZ_W);
  __bf16* Vb  = (__bf16*)(ws + 3 * SZ_X + 4 * SZ_W);
  __bf16* Vtb = (__bf16*)(ws + 4 * SZ_X + 4 * SZ_W);
  __bf16* Ctx = (__bf16*)(ws + 5 * SZ_X + 4 * SZ_W);

  const int nX = MROWS * HID;   // 8,388,608
  const int nW = HID * HID;     // 4,194,304

  // 1) fp32 -> bf16
  f32_to_bf16_k<<<nX / (256 * 8), 256, 0, stream>>>(X,  Xb,  nX);
  f32_to_bf16_k<<<nW / (256 * 8), 256, 0, stream>>>(Wq, Wqb, nW);
  f32_to_bf16_k<<<nW / (256 * 8), 256, 0, stream>>>(Wk, Wkb, nW);
  f32_to_bf16_k<<<nW / (256 * 8), 256, 0, stream>>>(Wv, Wvb, nW);
  f32_to_bf16_k<<<nW / (256 * 8), 256, 0, stream>>>(Wo, Wob, nW);

  // 2) projections; fold softmax scale * log2(e) into Q
  const float scale_q = 0.08838834764831845f /* 128^-0.5 */ * 1.4426950408889634f;
  const int gemm_blocks = (MROWS / 64) * (HID / 64) / 8;  // 256
  gemm_nt_bf16<0><<<gemm_blocks, 256, 0, stream>>>(Xb, Wqb, bq, Qb, MROWS, HID, HID, scale_q);
  gemm_nt_bf16<0><<<gemm_blocks, 256, 0, stream>>>(Xb, Wkb, bk, Kb, MROWS, HID, HID, 1.0f);
  gemm_nt_bf16<0><<<gemm_blocks, 256, 0, stream>>>(Xb, Wvb, bv, Vb, MROWS, HID, HID, 1.0f);

  // 3) V -> Vt [B,NH,HD,SEQ]
  transpose_v_k<<<nX / 256, 256, 0, stream>>>(Vb, Vtb);

  // 4) causal flash attention -> Ctx
  const int fa_blocks = BATCH * NH * (SEQ / 16) / 8;  // 512
  flash_attn_k<<<fa_blocks, 256, 0, stream>>>(Qb, Kb, Vtb, Ctx);

  // 5) output projection, fp32 out
  gemm_nt_bf16<1><<<gemm_blocks, 256, 0, stream>>>(Ctx, Wob, bo, out, MROWS, HID, HID, 1.0f);
}
